// ToMeCLIPEncoderLayer_53858889892481
// MI455X (gfx1250) — matrix operand order repaired and checked by reading, not compiled, not run on hardware
//
#include <hip/hip_runtime.h>
#include <hip/hip_bf16.h>

typedef _Float16 h16;
typedef __attribute__((ext_vector_type(16))) _Float16 v16h;
typedef __attribute__((ext_vector_type(8)))  _Float16 v8h;
typedef __attribute__((ext_vector_type(8)))  float    v8f;

typedef __attribute__((ext_vector_type(4))) int v4i;
typedef __attribute__((address_space(1))) v4i g_v4i;   // global int4
typedef __attribute__((address_space(3))) v4i l_v4i;   // LDS int4

#define B_   64
#define T_   257
#define D_   1024
#define H_   16
#define HD_  64
#define F_   4096
#define R_   16
#define TA_  129            // number of even rows (a-set)
#define UN_  (TA_ - R_)     // 113 unmerged
#define TP_  (T_ - R_)      // 241 tokens after merge
#define M1_  (B_ * T_)      // 16448
#define M2_  (B_ * TP_)     // 15424

static __device__ __forceinline__ v8f wmma_f16(v16h a, v16h b, v8f c) {
    return __builtin_amdgcn_wmma_f32_16x16x32_f16(false, a, false, b, (short)0, c, false, false);
}

// ---- CDNA5 async global->LDS copy (16B), tracked by ASYNCcnt -------------
static __device__ __forceinline__ void async_copy_b128(const void* gp, void* lp) {
#if __has_builtin(__builtin_amdgcn_global_load_async_to_lds_b128)
    __builtin_amdgcn_global_load_async_to_lds_b128(
        (g_v4i*)(uintptr_t)gp, (l_v4i*)(uint32_t)(uintptr_t)lp, 0, 0);
#else
    asm volatile("global_load_async_to_lds_b128 %0, %1, off"
                 :: "v"((uint32_t)(uintptr_t)lp), "v"(gp)
                 : "memory");
#endif
}

template <int N>
static __device__ __forceinline__ void wait_async() {
#if __has_builtin(__builtin_amdgcn_s_wait_asynccnt)
    __builtin_amdgcn_s_wait_asynccnt((unsigned short)N);
#else
    asm volatile("s_wait_asynccnt %0" :: "i"(N) : "memory");
#endif
}

// ---------------------------------------------------------------- f32 -> f16
__global__ __launch_bounds__(256) void cvt_f16_kernel(const float* __restrict__ in,
                                                      h16* __restrict__ out, int n) {
    int i = blockIdx.x * 256 + threadIdx.x;
    if (i < n) out[i] = (h16)in[i];
}

// ---------------------------------------------------------------- LayerNorm -> f16
__global__ __launch_bounds__(256) void ln_kernel(const float* __restrict__ x,
                                                 const float* __restrict__ g,
                                                 const float* __restrict__ bb,
                                                 h16* __restrict__ out) {
    const int row = blockIdx.x;
    const int tid = threadIdx.x;
    const float* xr = x + (size_t)row * D_;
    __shared__ float red[256];

    float s = 0.f;
    for (int i = tid; i < D_; i += 256) s += xr[i];
    red[tid] = s; __syncthreads();
    for (int o = 128; o > 0; o >>= 1) { if (tid < o) red[tid] += red[tid + o]; __syncthreads(); }
    float mu = red[0] / (float)D_;
    __syncthreads();

    float s2 = 0.f;
    for (int i = tid; i < D_; i += 256) { float d = xr[i] - mu; s2 += d * d; }
    red[tid] = s2; __syncthreads();
    for (int o = 128; o > 0; o >>= 1) { if (tid < o) red[tid] += red[tid + o]; __syncthreads(); }
    float var = red[0] / (float)D_;
    float rstd = rsqrtf(var + 1e-5f);

    h16* orow = out + (size_t)row * D_;
    for (int i = tid; i < D_; i += 256)
        orow[i] = (h16)((xr[i] - mu) * rstd * g[i] + bb[i]);
}

// ---------------------------------------------------------------- WMMA GEMM
// C(MxN) = A(MxK) * B(KxN), f16 in / f32 accumulate, double-buffered async LDS.
// epi 0: outH = (acc+bias)*scale       (f16)
// epi 1: outH = quickgelu(acc+bias)    (f16)
// epi 2: outF = acc+bias+resid         (f32)
#define BM 64
#define BN 128
#define BK 32
__global__ __launch_bounds__(256) void gemm_wmma_kernel(
    const h16* __restrict__ A, const h16* __restrict__ Bm,
    const float* __restrict__ bias, const float* __restrict__ resid,
    float* __restrict__ outF, h16* __restrict__ outH,
    int M, int N, int K, float scale, int epi) {
    __shared__ __attribute__((aligned(32))) h16 Al[2][BM][BK];
    __shared__ __attribute__((aligned(32))) h16 Bl[2][BK][BN];

    const int tid  = threadIdx.x;
    const int lane = tid & 31;
    const int wave = tid >> 5;
    const int wm   = wave & 3;    // 0..3 -> M offset 16*wm
    const int wn   = wave >> 2;   // 0..1 -> N offset 64*wn
    const int bm   = blockIdx.x * BM;
    const int bn   = blockIdx.y * BN;
    const int mrow = lane & 15;
    const int half = lane >> 4;
    const int kb   = half * 8;

    const int ar = tid >> 2, ac = (tid & 3) * 8;   // A staging: 16B / thread
    const int br = tid >> 3, bc = (tid & 7) * 16;  // B staging: 32B / thread

    v8f cc[4] = {};

    auto issue = [&](int buf, int kk) {
        async_copy_b128(&A[(size_t)(bm + ar) * K + kk + ac], &Al[buf][ar][ac]);
        async_copy_b128(&Bm[(size_t)(kk + br) * N + bn + bc], &Bl[buf][br][bc]);
        async_copy_b128(&Bm[(size_t)(kk + br) * N + bn + bc + 8], &Bl[buf][br][bc + 8]);
    };

    issue(0, 0);
    const int nsteps = K / BK;
    for (int s = 0; s < nsteps; ++s) {
        const int buf = s & 1;
        if (s + 1 < nsteps) {
            issue(buf ^ 1, (s + 1) * BK);   // overlap next tile copy with compute
            wait_async<3>();                // retire current buffer's 3 copies
        } else {
            wait_async<0>();
        }
        __syncthreads();

        v16h a;
        {
            v8h lo = *(const v8h*)&Al[buf][wm * 16 + mrow][kb];
            v8h hi = *(const v8h*)&Al[buf][wm * 16 + mrow][16 + kb];
#pragma unroll
            for (int i = 0; i < 8; ++i) { a[i] = lo[i]; a[8 + i] = hi[i]; }
        }
        const h16* brow = &Bl[buf][lane][wn * 64];
#pragma unroll
        for (int nc = 0; nc < 4; ++nc) {
            v16h bf;
#pragma unroll
            for (int e = 0; e < 16; ++e) {
                int kr = (e < 8) ? (kb + e) : (16 + kb + (e - 8));
                bf[e] = Bl[buf][kr][wn * 64 + nc * 16 + mrow];
            }
            cc[nc] = wmma_f16(a, bf, cc[nc]);
        }
        __syncthreads();
    }

    const int row0 = bm + wm * 16 + half * 8;
    const int col0 = bn + wn * 64 + mrow;
#pragma unroll
    for (int nc = 0; nc < 4; ++nc) {
        const int col = col0 + nc * 16;
        const float bv = bias ? bias[col] : 0.f;
#pragma unroll
        for (int r = 0; r < 8; ++r) {
            const int row = row0 + r;
            float val = cc[nc][r] + bv;
            if (epi == 0) {
                outH[(size_t)row * N + col] = (h16)(val * scale);
            } else if (epi == 1) {
                float gelu = val / (1.f + __expf(-1.702f * val));
                outH[(size_t)row * N + col] = (h16)gelu;
            } else {
                float rr = resid ? resid[(size_t)row * N + col] : 0.f;
                outF[(size_t)row * N + col] = val + rr;
            }
        }
    }
}

// ---------------------------------------------------------------- Flash attention
// One wave per (q-tile of 16, b*H+h). Streams 32-key super-tiles with online
// softmax; K/V double-buffered through async global->LDS copies.
__global__ __launch_bounds__(32) void attn_kernel(const h16* __restrict__ q,
                                                  const h16* __restrict__ k,
                                                  const h16* __restrict__ v,
                                                  h16* __restrict__ o) {
    const int qt   = blockIdx.x;          // 0..16
    const int bh   = blockIdx.y;
    const int b    = bh / H_;
    const int h    = bh % H_;
    const int lane = threadIdx.x;
    const int mrow = lane & 15;
    const int half = lane >> 4;
    const int kb   = half * 8;

    __shared__ __attribute__((aligned(32))) h16 Kt[2][32][64];
    __shared__ __attribute__((aligned(32))) h16 Vt[2][32][64];
    __shared__ __attribute__((aligned(32))) h16 Pt[16][32];

    // Q A-fragments (d 0..31 and d 32..63), 16-bit A 16x32 ISA layout
    int qrow = qt * 16 + mrow; if (qrow >= T_) qrow = T_ - 1;
    const h16* qp = q + ((size_t)(b * T_ + qrow)) * D_ + h * HD_;
    v16h aq0, aq1;
#pragma unroll
    for (int e = 0; e < 16; ++e) {
        int d = (e < 8) ? (kb + e) : (16 + kb + (e - 8));
        aq0[e] = qp[d];
        aq1[e] = qp[32 + d];
    }

    v8f acc[4] = {};
    float mrun[8], lrun[8];
#pragma unroll
    for (int r = 0; r < 8; ++r) { mrun[r] = -1e30f; lrun[r] = 0.f; }

    auto issueKV = [&](int buf, int k0) {
        int key = k0 + lane;
        int kc  = key < T_ ? key : T_ - 1;
        const h16* kp = k + ((size_t)(b * T_ + kc)) * D_ + h * HD_;
        const h16* vp = v + ((size_t)(b * T_ + kc)) * D_ + h * HD_;
#pragma unroll
        for (int i = 0; i < 8; ++i) {
            async_copy_b128(kp + i * 8, &Kt[buf][lane][i * 8]);
            async_copy_b128(vp + i * 8, &Vt[buf][lane][i * 8]);
        }
    };

    issueKV(0, 0);
    for (int kt = 0; kt < 9; ++kt) {
        const int k0  = kt * 32;
        const int buf = kt & 1;
        if (kt + 1 < 9) {
            issueKV(buf ^ 1, k0 + 32);  // prefetch next super-tile
            wait_async<32>();           // wait until at most 32 copies (16 per buffer) pending
        } else {
            wait_async<0>();
        }
        __syncthreads();

        // score fragments: B rows = contraction d, cols = keys
        v16h bkA, bkB, bkC, bkD;
#pragma unroll
        for (int e = 0; e < 16; ++e) {
            int d = (e < 8) ? (kb + e) : (16 + kb + (e - 8));
            bkA[e] = Kt[buf][mrow][d];           // keys 0..15,  d 0..31
            bkB[e] = Kt[buf][mrow][32 + d];      // keys 0..15,  d 32..63
            bkC[e] = Kt[buf][16 + mrow][d];      // keys 16..31, d 0..31
            bkD[e] = Kt[buf][16 + mrow][32 + d]; // keys 16..31, d 32..63
        }
        v8f s0 = {}, s1 = {};
        s0 = wmma_f16(aq0, bkA, s0); s0 = wmma_f16(aq1, bkB, s0);
        s1 = wmma_f16(aq0, bkC, s1); s1 = wmma_f16(aq1, bkD, s1);

        const int key0 = k0 + (lane & 15);
        const int key1 = key0 + 16;
#pragma unroll
        for (int r = 0; r < 8; ++r) {
            float a0 = (key0 < T_) ? s0[r] : -1e30f;
            float a1 = (key1 < T_) ? s1[r] : -1e30f;
            float tm = fmaxf(a0, a1);
            tm = fmaxf(tm, __shfl_xor(tm, 1));
            tm = fmaxf(tm, __shfl_xor(tm, 2));
            tm = fmaxf(tm, __shfl_xor(tm, 4));
            tm = fmaxf(tm, __shfl_xor(tm, 8));
            float nm    = fmaxf(mrun[r], tm);
            float alpha = __expf(mrun[r] - nm);
            float p0    = __expf(a0 - nm);
            float p1    = __expf(a1 - nm);
            float rs = p0 + p1;
            rs += __shfl_xor(rs, 1); rs += __shfl_xor(rs, 2);
            rs += __shfl_xor(rs, 4); rs += __shfl_xor(rs, 8);
            lrun[r] = lrun[r] * alpha + rs;
            mrun[r] = nm;
#pragma unroll
            for (int nc = 0; nc < 4; ++nc) acc[nc][r] *= alpha;
            Pt[r + 8 * half][lane & 15]        = (h16)p0;
            Pt[r + 8 * half][16 + (lane & 15)] = (h16)p1;
        }
        __syncthreads();

        // P as A-fragment (16x32), V as B-fragments (32 x 16-chunk of d)
        v16h ap;
        {
            v8h lo = *(const v8h*)&Pt[mrow][kb];
            v8h hi = *(const v8h*)&Pt[mrow][16 + kb];
#pragma unroll
            for (int i = 0; i < 8; ++i) { ap[i] = lo[i]; ap[8 + i] = hi[i]; }
        }
#pragma unroll
        for (int nc = 0; nc < 4; ++nc) {
            v16h bv;
#pragma unroll
            for (int e = 0; e < 16; ++e) {
                int kr = (e < 8) ? (kb + e) : (16 + kb + (e - 8));
                bv[e] = Vt[buf][kr][nc * 16 + mrow];
            }
            acc[nc] = wmma_f16(ap, bv, acc[nc]);
        }
        __syncthreads();
    }

#pragma unroll
    for (int r = 0; r < 8; ++r) {
        const int row = qt * 16 + r + 8 * half;
        if (row < T_) {
            float inv = 1.f / lrun[r];
            size_t base = ((size_t)(b * T_ + row)) * D_ + h * HD_ + (lane & 15);
#pragma unroll
            for (int nc = 0; nc < 4; ++nc)
                o[base + nc * 16] = (h16)(acc[nc][r] * inv);
        }
    }
}

// ---------------------------------------------------------------- row L2 norms
__global__ __launch_bounds__(256) void rownorm_kernel(const float* __restrict__ x,
                                                      float* __restrict__ norms) {
    const int row = blockIdx.x;
    const int tid = threadIdx.x;
    const float* xr = x + (size_t)row * D_;
    __shared__ float red[256];
    float s = 0.f;
    for (int i = tid; i < D_; i += 256) s += xr[i] * xr[i];
    red[tid] = s; __syncthreads();
    for (int o = 128; o > 0; o >>= 1) { if (tid < o) red[tid] += red[tid + o]; __syncthreads(); }
    if (tid == 0) norms[row] = sqrtf(red[0]);
}

// ---------------------------------------------------------------- bipartite match scores
__global__ __launch_bounds__(128) void match_kernel(const float* __restrict__ x1,
                                                    const float* __restrict__ norms,
                                                    float* __restrict__ node_max,
                                                    int* __restrict__ node_idx) {
    const int b = blockIdx.x;
    const int i = blockIdx.y;   // 0..128
    const int j = threadIdx.x;  // 0..127
    const float4* a4 = (const float4*)(x1 + ((size_t)(b * T_ + 2 * i)) * D_);
    const float4* b4 = (const float4*)(x1 + ((size_t)(b * T_ + 2 * j + 1)) * D_);
    float dot = 0.f;
    for (int d = 0; d < D_ / 4; ++d) {
        float4 av = a4[d], bv = b4[d];
        dot += av.x * bv.x + av.y * bv.y + av.z * bv.z + av.w * bv.w;
    }
    float s = dot / (norms[b * T_ + 2 * i] * norms[b * T_ + 2 * j + 1]);
    if (i == 0) s = -1e30f;  // class token row masked

    __shared__ float sv[128];
    __shared__ int   si[128];
    sv[j] = s; si[j] = j; __syncthreads();
    for (int o = 64; o > 0; o >>= 1) {
        if (j < o) {
            float other = sv[j + o]; int oi = si[j + o];
            if (other > sv[j] || (other == sv[j] && oi < si[j])) { sv[j] = other; si[j] = oi; }
        }
        __syncthreads();
    }
    if (j == 0) { node_max[b * TA_ + i] = sv[0]; node_idx[b * TA_ + i] = si[0]; }
}

// ---------------------------------------------------------------- stable rank sort per batch
__global__ void sort_kernel(const float* __restrict__ node_max,
                            const int* __restrict__ node_idx,
                            int* __restrict__ unm_idx, int* __restrict__ src_idx,
                            int* __restrict__ dst_idx) {
    const int b = blockIdx.x;
    if (threadIdx.x != 0) return;
    float key[TA_];
    int edge[TA_];
    for (int t = 0; t < TA_; ++t) key[t] = node_max[b * TA_ + t];
    // stable descending argsort (== jnp.argsort(-node_max)): rank by (key desc, idx asc)
    for (int t = 0; t < TA_; ++t) {
        int rank = 0;
        for (int u = 0; u < TA_; ++u) {
            if (key[u] > key[t] || (key[u] == key[t] && u < t)) rank++;
        }
        edge[rank] = t;
    }
    for (int s = 0; s < R_; ++s) {
        src_idx[b * R_ + s] = edge[s];
        dst_idx[b * R_ + s] = node_idx[b * TA_ + edge[s]];
    }
    // unm = edge[R:] sorted ascending (class_token path)
    int unm[UN_];
    for (int u = 0; u < UN_; ++u) unm[u] = edge[R_ + u];
    for (int u = 1; u < UN_; ++u) {
        int x = unm[u], p = u - 1;
        while (p >= 0 && unm[p] > x) { unm[p + 1] = unm[p]; --p; }
        unm[p + 1] = x;
    }
    for (int u = 0; u < UN_; ++u) unm_idx[b * UN_ + u] = unm[u];
}

// ---------------------------------------------------------------- token merge + size norm
__global__ __launch_bounds__(256) void merge_kernel(const float* __restrict__ x1,
                                                    const int* __restrict__ unm_idx,
                                                    const int* __restrict__ src_idx,
                                                    const int* __restrict__ dst_idx,
                                                    float* __restrict__ x2) {
    const int j = blockIdx.x;  // 0..240
    const int b = blockIdx.y;
    const int t = threadIdx.x;
    float* orow = x2 + ((size_t)(b * TP_ + j)) * D_;
    if (j < UN_) {
        int tok = 2 * unm_idx[b * UN_ + j];
        const float* src = x1 + ((size_t)(b * T_ + tok)) * D_;
        for (int d = t; d < D_; d += 256) orow[d] = src[d];
    } else {
        const int jd = j - UN_;
        int add[R_]; int na = 0; float cnt = 1.f;
        for (int s = 0; s < R_; ++s)
            if (dst_idx[b * R_ + s] == jd) { add[na++] = 2 * src_idx[b * R_ + s]; cnt += 1.f; }
        const float* dst = x1 + ((size_t)(b * T_ + 2 * jd + 1)) * D_;
        const float inv = 1.f / cnt;
        for (int d = t; d < D_; d += 256) {
            float a = dst[d];
            for (int s = 0; s < na; ++s) a += x1[((size_t)(b * T_ + add[s])) * D_ + d];
            orow[d] = a * inv;
        }
    }
}

// ================================================================ launch
extern "C" void kernel_launch(void* const* d_in, const int* in_sizes, int n_in,
                              void* d_out, int out_size, void* d_ws, size_t ws_size,
                              hipStream_t stream) {
    (void)in_sizes; (void)n_in; (void)out_size; (void)ws_size;
    const float* hidden = (const float*)d_in[0];
    const float* ln1_g  = (const float*)d_in[1];
    const float* ln1_b  = (const float*)d_in[2];
    const float* Wq = (const float*)d_in[3];  const float* bq = (const float*)d_in[4];
    const float* Wk = (const float*)d_in[5];  const float* bk = (const float*)d_in[6];
    const float* Wv = (const float*)d_in[7];  const float* bv = (const float*)d_in[8];
    const float* Wo = (const float*)d_in[9];  const float* bo = (const float*)d_in[10];
    const float* ln2_g = (const float*)d_in[11];
    const float* ln2_b = (const float*)d_in[12];
    const float* W1 = (const float*)d_in[13]; const float* b1 = (const float*)d_in[14];
    const float* W2 = (const float*)d_in[15]; const float* b2 = (const float*)d_in[16];

    char* ws = (char*)d_ws;
    size_t off = 0;
    auto alloc = [&](size_t bytes) -> void* {
        void* p = ws + off;
        off += (bytes + 255) & ~(size_t)255;
        return p;
    };
    h16* wq16   = (h16*)alloc((size_t)D_ * D_ * 2);
    h16* wk16   = (h16*)alloc((size_t)D_ * D_ * 2);
    h16* wv16   = (h16*)alloc((size_t)D_ * D_ * 2);
    h16* wo16   = (h16*)alloc((size_t)D_ * D_ * 2);
    h16* w1_16  = (h16*)alloc((size_t)D_ * F_ * 2);
    h16* w2_16  = (h16*)alloc((size_t)F_ * D_ * 2);
    h16* xln16  = (h16*)alloc((size_t)M1_ * D_ * 2);
    h16* q16    = (h16*)alloc((size_t)M1_ * D_ * 2);
    h16* k16    = (h16*)alloc((size_t)M1_ * D_ * 2);
    h16* v16    = (h16*)alloc((size_t)M1_ * D_ * 2);
    h16* attn16 = (h16*)alloc((size_t)M1_ * D_ * 2);
    h16* xln2   = (h16*)alloc((size_t)M2_ * D_ * 2);
    h16* h1     = (h16*)alloc((size_t)M2_ * F_ * 2);
    float* x1   = (float*)alloc((size_t)M1_ * D_ * 4);
    float* x2   = (float*)alloc((size_t)M2_ * D_ * 4);
    float* norms    = (float*)alloc((size_t)M1_ * 4);
    float* node_max = (float*)alloc((size_t)B_ * TA_ * 4);
    int*   node_idx = (int*)alloc((size_t)B_ * TA_ * 4);
    int*   unm_idx  = (int*)alloc((size_t)B_ * UN_ * 4);
    int*   src_idx  = (int*)alloc((size_t)B_ * R_ * 4);
    int*   dst_idx  = (int*)alloc((size_t)B_ * R_ * 4);

    auto cvt = [&](const float* src, h16* dst, int n) {
        cvt_f16_kernel<<<(n + 255) / 256, 256, 0, stream>>>(src, dst, n);
    };
    cvt(Wq, wq16, D_ * D_); cvt(Wk, wk16, D_ * D_);
    cvt(Wv, wv16, D_ * D_); cvt(Wo, wo16, D_ * D_);
    cvt(W1, w1_16, D_ * F_); cvt(W2, w2_16, F_ * D_);

    // LN1
    ln_kernel<<<M1_, 256, 0, stream>>>(hidden, ln1_g, ln1_b, xln16);

    // QKV projections (q pre-scaled by HD^-0.5)
    dim3 gD(M1_ / BM, D_ / BN);
    gemm_wmma_kernel<<<gD, 256, 0, stream>>>(xln16, wq16, bq, nullptr, nullptr, q16,
                                             M1_, D_, D_, 0.125f, 0);
    gemm_wmma_kernel<<<gD, 256, 0, stream>>>(xln16, wk16, bk, nullptr, nullptr, k16,
                                             M1_, D_, D_, 1.f, 0);
    gemm_wmma_kernel<<<gD, 256, 0, stream>>>(xln16, wv16, bv, nullptr, nullptr, v16,
                                             M1_, D_, D_, 1.f, 0);

    // attention
    attn_kernel<<<dim3(17, B_ * H_), 32, 0, stream>>>(q16, k16, v16, attn16);

    // O projection + residual -> x1 (f32)
    gemm_wmma_kernel<<<gD, 256, 0, stream>>>(attn16, wo16, bo, hidden, x1, nullptr,
                                             M1_, D_, D_, 1.f, 2);

    // token matching + merge
    rownorm_kernel<<<M1_, 256, 0, stream>>>(x1, norms);
    match_kernel<<<dim3(B_, TA_), 128, 0, stream>>>(x1, norms, node_max, node_idx);
    sort_kernel<<<B_, 32, 0, stream>>>(node_max, node_idx, unm_idx, src_idx, dst_idx);
    merge_kernel<<<dim3(TP_, B_), 256, 0, stream>>>(x1, unm_idx, src_idx, dst_idx, x2);

    // LN2 + MLP + residual -> d_out
    ln_kernel<<<M2_, 256, 0, stream>>>(x2, ln2_g, ln2_b, xln2);
    dim3 gF(M2_ / BM, F_ / BN);
    gemm_wmma_kernel<<<gF, 256, 0, stream>>>(xln2, w1_16, b1, nullptr, nullptr, h1,
                                             M2_, F_, D_, 1.f, 1);
    dim3 gO(M2_ / BM, D_ / BN);
    gemm_wmma_kernel<<<gO, 256, 0, stream>>>(h1, w2_16, b2, x2, (float*)d_out, nullptr,
                                             M2_, D_, F_, 1.f, 2);
}
